// MoE_84533546320329
// MI455X (gfx1250) — matrix operand order repaired and checked by reading, not compiled, not run on hardware
//
#include <hip/hip_runtime.h>
#include <hip/hip_bf16.h>

// MoE (top-2 of 8 experts + shared expert), a=2048 tokens, D=1024, F=2048.
// fp32 weights are converted once to bf16 in a [n][k] (k-contiguous) layout,
// then all GEMMs run on v_wmma_f32_16x16x32_bf16 with f32 accumulation.
// Block tile 64(M)x128(N): 8 waves, each wave holds 4 M-sub-tile accumulators
// and reuses each 32B B-fragment 4x. SwiGLU fused into GEMM1 epilogue;
// expert-sum + shared combine folded into GEMM2's K loop (K = 9*F).
// Workspace: ~189 MB.

typedef __bf16 bf16;
typedef bf16  v16bf __attribute__((ext_vector_type(16)));
typedef bf16  v8bf  __attribute__((ext_vector_type(8)));
typedef float v8f   __attribute__((ext_vector_type(8)));

#define NA   2048   // tokens (B*S)
#define ND   1024   // model dim
#define NF   2048   // expert hidden dim
#define NE   8      // experts
#define NSRC 9      // experts + shared

__device__ __forceinline__ float fast_sigmoid(float v) {
  // v_exp_f32 + v_add + v_rcp_f32 (avoid the IEEE-division expansion)
  return __builtin_amdgcn_rcpf(1.f + __expf(-v));
}

// ---------------------------------------------------------------- router ----
__global__ void moe_router(const float* __restrict__ x,
                           const float* __restrict__ rw,   // [D][E]
                           float* __restrict__ gates) {    // [E][a]
  const int t   = blockIdx.x;      // token
  const int tid = threadIdx.x;     // 128 threads
  float acc[NE];
#pragma unroll
  for (int e = 0; e < NE; ++e) acc[e] = 0.f;
  const float* xr = x + (size_t)t * ND;
  for (int d = tid; d < ND; d += 128) {
    const float xv = xr[d];
    const float* r = rw + (size_t)d * NE;
#pragma unroll
    for (int e = 0; e < NE; ++e) acc[e] += xv * r[e];
  }
  __shared__ float red[128][NE];
#pragma unroll
  for (int e = 0; e < NE; ++e) red[tid][e] = acc[e];
  __syncthreads();
  for (int s = 64; s > 0; s >>= 1) {
    if (tid < s) {
#pragma unroll
      for (int e = 0; e < NE; ++e) red[tid][e] += red[tid + s][e];
    }
    __syncthreads();
  }
  if (tid == 0) {
    float sc[NE];
#pragma unroll
    for (int e = 0; e < NE; ++e) sc[e] = red[0][e];
    int i1 = 0;
#pragma unroll
    for (int e = 1; e < NE; ++e) if (sc[e] > sc[i1]) i1 = e;
    int i2 = (i1 == 0) ? 1 : 0;
#pragma unroll
    for (int e = 0; e < NE; ++e) if (e != i1 && sc[e] > sc[i2]) i2 = e;
#pragma unroll
    for (int e = 0; e < NE; ++e) {
      const float g = (e == i1 || e == i2) ? fast_sigmoid(sc[e]) : 0.f;
      gates[(size_t)e * NA + t] = g;
    }
  }
}

// ------------------------------------------- weight convert / transpose -----
// out[src][n][k] bf16 (k contiguous).  experts: we[src][k][n]  (LDS transpose)
//                                      shared : ws[n][k]       (direct copy)
__global__ void moe_convert_w(const float* __restrict__ we,
                              const float* __restrict__ ws,
                              bf16* __restrict__ out, int NN, int KK) {
  const int src = blockIdx.z;
  const int n0 = blockIdx.x * 32, k0 = blockIdx.y * 32;
  const int tx = threadIdx.x & 31, ty = threadIdx.x >> 5;   // 32 x 8
  if (src < NE) {
    __shared__ float tile[32][33];
    const float* base = we + (size_t)src * KK * NN;
#pragma unroll
    for (int r = ty; r < 32; r += 8)
      tile[r][tx] = base[(size_t)(k0 + r) * NN + (n0 + tx)];   // tile[k][n]
    __syncthreads();
#pragma unroll
    for (int r = ty; r < 32; r += 8)
      out[((size_t)src * NN + n0 + r) * KK + (k0 + tx)] = (bf16)tile[tx][r];
  } else {
#pragma unroll
    for (int r = ty; r < 32; r += 8)
      out[((size_t)NE * NN + n0 + r) * KK + (k0 + tx)] =
          (bf16)ws[(size_t)(n0 + r) * KK + (k0 + tx)];
  }
}

// --------------------------------------------------------------- helpers ----
__device__ __forceinline__ v16bf load_a_frag(const bf16* __restrict__ As,
                                             int mrow, int g16) {
  // A 16x32 bf16: elems 0..15 = K[g*16 .. g*16+15] (contiguous, matches B)
  const v8bf alo = *(const v8bf*)(As + mrow * 32 + g16 * 16);
  const v8bf ahi = *(const v8bf*)(As + mrow * 32 + g16 * 16 + 8);
  v16bf a;
#pragma unroll
  for (int i = 0; i < 8; ++i) { a[i] = alo[i]; a[i + 8] = ahi[i]; }
  return a;
}

__device__ __forceinline__ unsigned int pack2bf(float a, float b) {
  const bf16 lo = (bf16)a, hi = (bf16)b;
  const unsigned short ul = __builtin_bit_cast(unsigned short, lo);
  const unsigned short uh = __builtin_bit_cast(unsigned short, hi);
  return ((unsigned int)uh << 16) | ul;
}

__device__ __forceinline__ v8f wmma_bf16(v16bf a, v16bf b, v8f c) {
  return __builtin_amdgcn_wmma_f32_16x16x32_bf16(false, a, false, b,
                                                 (short)0, c, false, false);
}

// ------------------------------------------------- GEMM1 + fused SwiGLU -----
// h[src][m][n] = silu(Xg @ W1t[src]^T) * (Xg @ W3t[src]^T), bf16 out
__global__ void
__launch_bounds__(256)
moe_gemm1(const float* __restrict__ x, const float* __restrict__ gates,
          const bf16* __restrict__ W1t, const bf16* __restrict__ W3t,
          bf16* __restrict__ h) {
  const int src  = blockIdx.z;
  const int M0   = blockIdx.y * 64;
  const int tid  = threadIdx.x;
  const int wave = tid >> 5, lane = tid & 31;
  const int N0   = blockIdx.x * 128 + wave * 16;
  const int g16  = lane >> 4, nlane = lane & 15;

  __shared__ __align__(32) bf16 As[64 * 32];        // 4 KB
  unsigned int* AsU = (unsigned int*)As;

  // A-staging plan: 64x32 tile = 1024 dwords; thread does 4 packed dwords.
  // Gates are row-only -> hoisted out of the K loop (branch-free inner loop).
  const float* xp[4];
  float gg[4];
#pragma unroll
  for (int p = 0; p < 4; ++p) {
    const int e2 = tid + p * 256;
    const int r  = e2 >> 4;            // row 0..63
    const int cc = (e2 & 15) * 2;      // col 0,2,..,30
    xp[p] = x + (size_t)(M0 + r) * ND + cc;
    gg[p] = (src < NE) ? gates[(size_t)src * NA + M0 + r] : 1.f;
  }

  // B fragment: lane holds N = lane&15, K = g16*16 + i (16 contiguous bf16)
  const size_t boff = ((size_t)src * NF + N0 + nlane) * (size_t)ND + g16 * 16;
  const bf16* b1p = W1t + boff;
  const bf16* b3p = W3t + boff;

  v8f c1[4] = {};
  v8f c3[4] = {};
  for (int k0 = 0; k0 < ND; k0 += 32) {
    __syncthreads();
#pragma unroll
    for (int p = 0; p < 4; ++p) {
      const float2 xv = *(const float2*)(xp[p] + k0);
      AsU[tid + p * 256] = pack2bf(xv.x * gg[p], xv.y * gg[p]);
    }
    __syncthreads();

    const v16bf b1 = *(const v16bf*)(b1p + k0);
    const v16bf b3 = *(const v16bf*)(b3p + k0);
    __builtin_prefetch(b1p + k0 + 32, 0, 1);
    __builtin_prefetch(b3p + k0 + 32, 0, 1);
#pragma unroll
    for (int m = 0; m < 4; ++m) {
      const v16bf a = load_a_frag(As + m * 16 * 32, nlane, g16);
      c1[m] = wmma_bf16(a, b1, c1[m]);
      c3[m] = wmma_bf16(a, b3, c3[m]);
    }
  }

  // epilogue: h = silu(c1) * c3    (C/D: N = lane&15, M = v + 8*(lane>>4))
  const int n = N0 + nlane;
#pragma unroll
  for (int m = 0; m < 4; ++m) {
#pragma unroll
    for (int v = 0; v < 8; ++v) {
      const int mm = M0 + m * 16 + v + 8 * g16;
      const float mid = c1[m][v];
      const float hv  = mid * fast_sigmoid(mid) * c3[m][v];
      h[((size_t)src * NA + mm) * NF + n] = (bf16)hv;
    }
  }
}

// ------------------------------------ GEMM2: combine experts + shared -------
// out[m][n] = sum_src sum_f h[src][m][f] * W2t[src][n][f]
__global__ void
__launch_bounds__(256)
moe_gemm2(const bf16* __restrict__ h, const bf16* __restrict__ W2t,
          float* __restrict__ out) {
  const int M0   = blockIdx.y * 64;
  const int tid  = threadIdx.x;
  const int wave = tid >> 5, lane = tid & 31;
  const int N0   = blockIdx.x * 128 + wave * 16;
  const int g16  = lane >> 4, nlane = lane & 15;

  __shared__ __align__(32) bf16 As[64 * 32];        // 4 KB
  unsigned int* AsU = (unsigned int*)As;

  int rr[4], cc[4];
#pragma unroll
  for (int p = 0; p < 4; ++p) {
    const int e2 = tid + p * 256;
    rr[p] = e2 >> 4;                 // row 0..63
    cc[p] = (e2 & 15) * 2;           // col (bf16 units)
  }

  v8f c[4] = {};
  for (int src = 0; src < NSRC; ++src) {
    const bf16* bp = W2t + ((size_t)src * ND + N0 + nlane) * (size_t)NF
                         + g16 * 16;
    const bf16* hbase = h + ((size_t)src * NA + M0) * (size_t)NF;
    for (int k0 = 0; k0 < NF; k0 += 32) {
      __syncthreads();
#pragma unroll
      for (int p = 0; p < 4; ++p)
        AsU[tid + p * 256] =
            *(const unsigned int*)(hbase + (size_t)rr[p] * NF + k0 + cc[p]);
      __syncthreads();

      const v16bf b = *(const v16bf*)(bp + k0);
      __builtin_prefetch(bp + k0 + 32, 0, 1);
#pragma unroll
      for (int m = 0; m < 4; ++m) {
        const v16bf a = load_a_frag(As + m * 16 * 32, nlane, g16);
        c[m] = wmma_bf16(a, b, c[m]);
      }
    }
  }

  const int n = N0 + nlane;
#pragma unroll
  for (int m = 0; m < 4; ++m) {
#pragma unroll
    for (int v = 0; v < 8; ++v) {
      const int mm = M0 + m * 16 + v + 8 * g16;
      out[(size_t)mm * ND + n] = c[m][v];
    }
  }
}

// ---------------------------------------------------------------- launch ----
extern "C" void kernel_launch(void* const* d_in, const int* in_sizes, int n_in,
                              void* d_out, int out_size, void* d_ws,
                              size_t ws_size, hipStream_t stream) {
  const float* x        = (const float*)d_in[0];  // [2,1024,1024]
  const float* router   = (const float*)d_in[1];  // [1024,8]
  const float* w_in_sh  = (const float*)d_in[2];  // [F,D]
  const float* w_out_sh = (const float*)d_in[3];  // [D,F]
  const float* w_sw_sh  = (const float*)d_in[4];  // [F,D]
  const float* w_in_e   = (const float*)d_in[5];  // [E,D,F]
  const float* w_sw_e   = (const float*)d_in[6];  // [E,D,F]
  const float* w_out_e  = (const float*)d_in[7];  // [E,F,D]
  float* out = (float*)d_out;

  char* ws = (char*)d_ws;
  const size_t szGates = (size_t)NE * NA * sizeof(float);            // 64 KB
  const size_t szW1    = (size_t)NSRC * NF * ND * sizeof(bf16);      // 37.7 MB
  const size_t szW2    = (size_t)NSRC * ND * NF * sizeof(bf16);      // 37.7 MB
  float* gates = (float*)(ws);
  bf16*  W1t   = (bf16*)(ws + szGates);
  bf16*  W3t   = (bf16*)(ws + szGates + szW1);
  bf16*  W2t   = (bf16*)(ws + szGates + 2 * szW1);
  bf16*  h     = (bf16*)(ws + szGates + 2 * szW1 + szW2);            // 75.5 MB

  // 1) router gates
  moe_router<<<NA, 128, 0, stream>>>(x, router, gates);

  // 2) weight convert (fp32 -> bf16, [n][k] k-contiguous layout)
  moe_convert_w<<<dim3(NF / 32, ND / 32, NSRC), 256, 0, stream>>>(
      w_in_e, w_in_sh, W1t, NF, ND);
  moe_convert_w<<<dim3(NF / 32, ND / 32, NSRC), 256, 0, stream>>>(
      w_sw_e, w_sw_sh, W3t, NF, ND);
  moe_convert_w<<<dim3(ND / 32, NF / 32, NSRC), 256, 0, stream>>>(
      w_out_e, w_out_sh, W2t, ND, NF);

  // 3) GEMM1 + fused SwiGLU  -> h[9][a][F]
  moe_gemm1<<<dim3(NF / 128, NA / 64, NSRC), 256, 0, stream>>>(
      x, gates, W1t, W3t, h);

  // 4) GEMM2 over K = 9*F, writes final fp32 output
  moe_gemm2<<<dim3(ND / 128, NA / 64), 256, 0, stream>>>(h, W2t, out);
}
